// GraphTransformerEncoderLayer_46437186404983
// MI455X (gfx1250) — matrix operand order repaired and checked by reading, not compiled, not run on hardware
//
#include <hip/hip_runtime.h>
#include <hip/hip_bf16.h>
#include <math.h>

typedef _Float16 v16h __attribute__((ext_vector_type(16)));
typedef _Float16 v8h  __attribute__((ext_vector_type(8)));
typedef float    v8f  __attribute__((ext_vector_type(8)));

#define CDIV(a,b) (((a)+(b)-1)/(b))

__device__ __forceinline__ unsigned f2ord(float f) {
    unsigned u = __float_as_uint(f);
    return (u & 0x80000000u) ? ~u : (u | 0x80000000u);
}
__device__ __forceinline__ float ord2f(unsigned u) {
    return __uint_as_float((u & 0x80000000u) ? (u & 0x7fffffffu) : ~u);
}

// ---------------- weight pack: W[K][Ncols] f32 -> Wt[Ncols][K] f16 -------------
__global__ void pack_wt(const float* __restrict__ W, _Float16* __restrict__ Wt,
                        int K, int Ncols) {
    int i = blockIdx.x * 256 + threadIdx.x;
    if (i >= K * Ncols) return;
    int col = i / K, k = i - col * K;
    Wt[i] = (_Float16)W[(size_t)k * Ncols + col];
}

__global__ void build_bcat(const float* __restrict__ bq, const float* __restrict__ bk,
                           const float* __restrict__ bv, float* __restrict__ bcat) {
    int i = blockIdx.x * 256 + threadIdx.x;
    if (i >= 768) return;
    bcat[i] = (i < 256) ? bq[i] : ((i < 512) ? bk[i - 256] : bv[i - 512]);
}

__global__ void f2h_kernel(const float* __restrict__ s, _Float16* __restrict__ d, int n) {
    int i = blockIdx.x * 256 + threadIdx.x;
    if (i < n) d[i] = (_Float16)s[i];
}

// ---------------- WMMA GEMM: C[M][Ncols] = A[M][K](f16) @ Bt[Ncols][K]^T + bias -
// B panel (64 cols x K) is staged once per block into LDS with CDNA5 async
// global->LDS copies (ASYNCcnt), then consumed via ds_load per K-step.
// MODE 0: f32 store; MODE 1: f16 store; MODE 2: exact-gelu epilogue + f16 store
template <int MODE>
__global__ __launch_bounds__(256) void wmma_gemm(
        const _Float16* __restrict__ A, const _Float16* __restrict__ Bt,
        const float* __restrict__ bias, void* __restrict__ Cout,
        int M, int Ncols, int K, int ldc) {
    // padded row stride (halves): +8 halves = +16B so consecutive columns land on
    // different LDS banks (stride 272B -> 4-bank advance per column)
    const int ldb = K + 8;
    __shared__ _Float16 Bs[64 * 264];      // supports K up to 256

    const int lane   = threadIdx.x & 31;
    const int wave   = threadIdx.x >> 5;
    const int lane16 = lane & 15;
    const int laneHi = lane >> 4;
    const int rowBase = blockIdx.y * 128 + wave * 16;
    const int colBase = blockIdx.x * 64;

    // ---- async stage of B panel: LDS[c*ldb + k] = Bt[(colBase+c)*K + k] ----
    {
        const int chunks = (64 * K) / 8;                 // 16B chunks
        const int perCol = K / 8;
        for (int ch = threadIdx.x; ch < chunks; ch += 256) {
            int c  = ch / perCol;
            int k8 = (ch - c * perCol) * 8;
            // low 32 bits of a flat pointer into __shared__ == LDS byte offset
            unsigned lds = (unsigned)(size_t)(&Bs[c * ldb + k8]);
            unsigned long long ga =
                (unsigned long long)(size_t)(Bt + (size_t)(colBase + c) * K + k8);
            asm volatile("global_load_async_to_lds_b128 %0, %1, off"
                         :: "v"(lds), "v"(ga) : "memory");
        }
        asm volatile("s_wait_asynccnt 0x0" ::: "memory");
        __syncthreads();
    }

    int arow = rowBase + lane16;
    if (arow > M - 1) arow = M - 1;        // clamp; stores are guarded
    const _Float16* Arow = A + (size_t)arow * K;

    v8f acc[4] = {v8f{}, v8f{}, v8f{}, v8f{}};

    for (int k0 = 0; k0 < K; k0 += 32) {
        // A fragment: 16-bit A 16x32 layout (ISA 7.12.2): lanes0-15 K=[k0..k0+7],[k0+16..k0+23]
        //             lanes16-31 K=[k0+8..],[k0+24..]
        int kA = k0 + laneHi * 8;
        v8h a0 = *(const v8h*)(Arow + kA);
        v8h a1 = *(const v8h*)(Arow + kA + 16);
        v16h a = __builtin_shufflevector(a0, a1, 0,1,2,3,4,5,6,7,8,9,10,11,12,13,14,15);

        // B fragment from LDS: lanes0-15 hold K=[k0..k0+7],[k0+16..k0+23] of column lane16,
        //                      lanes16-31 hold K=[k0+8..],[k0+24..]
        int kB = k0 + laneHi * 8;
#pragma unroll
        for (int t = 0; t < 4; t++) {
            const _Float16* Bp = &Bs[(lane16 + 16 * t) * ldb + kB];
            v8h b0 = *(const v8h*)(Bp);
            v8h b1 = *(const v8h*)(Bp + 16);
            v16h b = __builtin_shufflevector(b0, b1, 0,1,2,3,4,5,6,7,8,9,10,11,12,13,14,15);
            acc[t] = __builtin_amdgcn_wmma_f32_16x16x32_f16(
                false, a, false, b, (short)0, acc[t], false, false);
        }
    }

#pragma unroll
    for (int t = 0; t < 4; t++) {
        int col = colBase + t * 16 + lane16;
        float bs = bias[col];
#pragma unroll
        for (int r = 0; r < 8; r++) {
            int row = rowBase + laneHi * 8 + r;   // C/D layout: VGPR r -> M=r (+8 for hi half)
            if (row < M) {
                float v = acc[t][r] + bs;
                if constexpr (MODE == 2)
                    v = 0.5f * v * (1.0f + erff(v * 0.70710678118654752f));
                if constexpr (MODE == 0)
                    ((float*)Cout)[(size_t)row * ldc + col] = v;
                else
                    ((_Float16*)Cout)[(size_t)row * ldc + col] = (_Float16)v;
            }
        }
    }
}

// ---------------- per-edge attention logits + segment max -----------------------
// qkv row layout: [q(256) | k(256) | v(256)] f16, stride 768. One wave per edge;
// lanes 0-15 -> head 0, lanes 16-31 -> head 1 (8 halves each).
__global__ __launch_bounds__(256) void alpha_kernel(
        const _Float16* __restrict__ qkv, const int* __restrict__ ei,
        float* __restrict__ alpha, unsigned* __restrict__ mmax, int E) {
    int e = blockIdx.x * 8 + (threadIdx.x >> 5);
    if (e >= E) return;
    int lane = threadIdx.x & 31;
    int src = ei[e], dst = ei[E + e];
    const _Float16* q = qkv + (size_t)dst * 768;
    const _Float16* k = qkv + (size_t)src * 768 + 256;
    int j = lane * 8;
    v8h qv = *(const v8h*)(q + j);
    v8h kv = *(const v8h*)(k + j);
    float s = 0.f;
#pragma unroll
    for (int i = 0; i < 8; i++) s += (float)qv[i] * (float)kv[i];
#pragma unroll
    for (int m = 1; m < 16; m <<= 1) s += __shfl_xor(s, m, 32);
    if ((lane & 15) == 0) {
        int hh = lane >> 4;
        float a = s * 0.08838834764831845f;   // 1/sqrt(128)
        alpha[(size_t)e * 2 + hh] = a;
        atomicMax(mmax + (size_t)dst * 2 + hh, f2ord(a));
    }
}

// ---------------- exp(alpha - m) in-place + segment denom -----------------------
__global__ void ex_kernel(float* __restrict__ alpha, const int* __restrict__ ei,
                          const unsigned* __restrict__ mmax, float* __restrict__ den, int E) {
    int i = blockIdx.x * 256 + threadIdx.x;
    if (i >= E * 2) return;
    int e = i >> 1, hh = i & 1;
    int dst = ei[E + e];
    float m = ord2f(mmax[(size_t)dst * 2 + hh]);
    float ex = __expf(alpha[i] - m);
    alpha[i] = ex;
    atomicAdd(den + (size_t)dst * 2 + hh, ex);
}

// ---------------- weighted scatter: agg[dst][h][c] += ex * v[src][h][c] ---------
__global__ __launch_bounds__(256) void agg_kernel(
        const _Float16* __restrict__ qkv, const int* __restrict__ ei,
        const float* __restrict__ attn, float* __restrict__ agg, int E) {
    int e = blockIdx.x * 8 + (threadIdx.x >> 5);
    if (e >= E) return;
    int lane = threadIdx.x & 31;
    int src = ei[e], dst = ei[E + e];
    int j = lane * 8;                 // 0..255, head = j>>7 (no 8-chunk crosses c=128)
    float w = attn[(size_t)e * 2 + (j >> 7)];
    v8h vv = *(const v8h*)(qkv + (size_t)src * 768 + 512 + j);
    float* ap = agg + (size_t)dst * 256 + j;
#pragma unroll
    for (int i = 0; i < 8; i++) atomicAdd(ap + i, (float)vv[i] * w);
}

// ---------------- combine heads + skip + residual + LayerNorm1 ------------------
__global__ __launch_bounds__(256) void combine_ln1(
        const float* __restrict__ x, const float* __restrict__ agg,
        const float* __restrict__ den, const float* __restrict__ skip,
        const float* __restrict__ g1, const float* __restrict__ be1,
        float* __restrict__ h, _Float16* __restrict__ hh, int Nn) {
    int n = blockIdx.x * 8 + (threadIdx.x >> 5);
    if (n >= Nn) return;
    int lane = threadIdx.x & 31;
    float inv0 = 0.5f / (den[(size_t)n * 2 + 0] + 1e-16f);
    float inv1 = 0.5f / (den[(size_t)n * 2 + 1] + 1e-16f);
    const float* a0 = agg + (size_t)n * 256;
    const float* xr = x + (size_t)n * 128;
    const float* sr = skip + (size_t)n * 128;
    float t[4], s = 0.f, ss = 0.f;
#pragma unroll
    for (int r = 0; r < 4; r++) {
        int c = lane * 4 + r;
        float tv = xr[c] + a0[c] * inv0 + a0[128 + c] * inv1 + sr[c];
        t[r] = tv; s += tv; ss += tv * tv;
    }
#pragma unroll
    for (int m = 1; m < 32; m <<= 1) { s += __shfl_xor(s, m, 32); ss += __shfl_xor(ss, m, 32); }
    float mu = s * 0.0078125f;
    float var = ss * 0.0078125f - mu * mu;
    float rs = rsqrtf(var + 1e-5f);
    float* hr = h + (size_t)n * 128;
    _Float16* hhr = hh + (size_t)n * 128;
#pragma unroll
    for (int r = 0; r < 4; r++) {
        int c = lane * 4 + r;
        float hv = (t[r] - mu) * rs * g1[c] + be1[c];
        hr[c] = hv; hhr[c] = (_Float16)hv;
    }
}

// ---------------- residual + LayerNorm2 -> d_out --------------------------------
__global__ __launch_bounds__(256) void combine_ln2(
        const float* __restrict__ h, const float* __restrict__ ff,
        const float* __restrict__ g2, const float* __restrict__ be2,
        float* __restrict__ out, int Nn) {
    int n = blockIdx.x * 8 + (threadIdx.x >> 5);
    if (n >= Nn) return;
    int lane = threadIdx.x & 31;
    const float* hr = h + (size_t)n * 128;
    const float* fr = ff + (size_t)n * 128;
    float t[4], s = 0.f, ss = 0.f;
#pragma unroll
    for (int r = 0; r < 4; r++) {
        int c = lane * 4 + r;
        float tv = hr[c] + fr[c];
        t[r] = tv; s += tv; ss += tv * tv;
    }
#pragma unroll
    for (int m = 1; m < 32; m <<= 1) { s += __shfl_xor(s, m, 32); ss += __shfl_xor(ss, m, 32); }
    float mu = s * 0.0078125f;
    float var = ss * 0.0078125f - mu * mu;
    float rs = rsqrtf(var + 1e-5f);
    float* orow = out + (size_t)n * 128;
#pragma unroll
    for (int r = 0; r < 4; r++) {
        int c = lane * 4 + r;
        orow[c] = (t[r] - mu) * rs * g2[c] + be2[c];
    }
}

extern "C" void kernel_launch(void* const* d_in, const int* in_sizes, int n_in,
                              void* d_out, int out_size, void* d_ws, size_t ws_size,
                              hipStream_t stream) {
    const float* x     = (const float*)d_in[0];
    const int*   ei    = (const int*)d_in[1];
    const float* Wq    = (const float*)d_in[2];
    const float* bq    = (const float*)d_in[3];
    const float* Wk    = (const float*)d_in[4];
    const float* bk    = (const float*)d_in[5];
    const float* Wv    = (const float*)d_in[6];
    const float* bv    = (const float*)d_in[7];
    const float* Wskip = (const float*)d_in[8];
    const float* bskip = (const float*)d_in[9];
    const float* g1    = (const float*)d_in[10];
    const float* be1   = (const float*)d_in[11];
    const float* W1    = (const float*)d_in[12];
    const float* b1    = (const float*)d_in[13];
    const float* W2    = (const float*)d_in[14];
    const float* b2    = (const float*)d_in[15];
    const float* g2    = (const float*)d_in[16];
    const float* be2   = (const float*)d_in[17];

    const int N = in_sizes[0] / 128;
    const int E = in_sizes[1] / 2;

    // ---- workspace layout (lifetimes allow the aliases noted below) ----
    char* base = (char*)d_ws;
    size_t off = 0;
    auto alloc = [&](size_t b) { size_t o = off; off += (b + 255) & ~(size_t)255; return o; };
    _Float16* xh    = (_Float16*)(base + alloc((size_t)N * 128 * 2));
    _Float16* qkv   = (_Float16*)(base + alloc((size_t)N * 768 * 2));
    float*    skipf = (float*)   (base + alloc((size_t)N * 128 * 4));
    float*    alpha = (float*)   (base + alloc((size_t)E * 2 * 4));
    unsigned* mmax  = (unsigned*)(base + alloc((size_t)N * 2 * 4));
    float*    den   = (float*)   (base + alloc((size_t)N * 2 * 4));
    float*    agg   = (float*)   (base + alloc((size_t)N * 256 * 4));
    float*    hbuf  = (float*)   (base + alloc((size_t)N * 128 * 4));
    _Float16* wts   = (_Float16*)(base + alloc((size_t)768 * 128 * 2));
    float*    bcat  = (float*)   (base + alloc((size_t)768 * 4));
    _Float16* wskt  = (_Float16*)(base + alloc((size_t)128 * 128 * 2));
    _Float16* w1t   = (_Float16*)(base + alloc((size_t)256 * 128 * 2));
    _Float16* w2t   = (_Float16*)(base + alloc((size_t)128 * 256 * 2));
    _Float16* hh  = xh;     // xh dead after the two input GEMMs
    _Float16* ffa = qkv;    // qkv dead after agg_kernel
    float*    ff2 = skipf;  // skipf dead after combine_ln1

    // ---- weight pack / convert ----
    pack_wt<<<CDIV(128 * 256, 256), 256, 0, stream>>>(Wq, wts, 128, 256);
    pack_wt<<<CDIV(128 * 256, 256), 256, 0, stream>>>(Wk, wts + 256 * 128, 128, 256);
    pack_wt<<<CDIV(128 * 256, 256), 256, 0, stream>>>(Wv, wts + 512 * 128, 128, 256);
    pack_wt<<<CDIV(128 * 128, 256), 256, 0, stream>>>(Wskip, wskt, 128, 128);
    pack_wt<<<CDIV(128 * 256, 256), 256, 0, stream>>>(W1, w1t, 128, 256);
    pack_wt<<<CDIV(256 * 128, 256), 256, 0, stream>>>(W2, w2t, 256, 128);
    build_bcat<<<3, 256, 0, stream>>>(bq, bk, bv, bcat);
    f2h_kernel<<<CDIV(N * 128, 256), 256, 0, stream>>>(x, xh, N * 128);

    // ---- input GEMMs (WMMA f16 -> f32 acc, async-LDS B staging) ----
    wmma_gemm<1><<<dim3(768 / 64, CDIV(N, 128)), 256, 0, stream>>>(xh, wts, bcat, qkv, N, 768, 128, 768);
    wmma_gemm<0><<<dim3(128 / 64, CDIV(N, 128)), 256, 0, stream>>>(xh, wskt, bskip, skipf, N, 128, 128, 128);

    // ---- edge attention ----
    (void)hipMemsetAsync(mmax, 0, (size_t)N * 2 * 4, stream);   // 0 == ordered(-inf) floor
    (void)hipMemsetAsync(den, 0, (size_t)N * 2 * 4, stream);
    (void)hipMemsetAsync(agg, 0, (size_t)N * 256 * 4, stream);
    alpha_kernel<<<CDIV(E, 8), 256, 0, stream>>>(qkv, ei, alpha, mmax, E);
    ex_kernel<<<CDIV(E * 2, 256), 256, 0, stream>>>(alpha, ei, mmax, den, E);
    agg_kernel<<<CDIV(E, 8), 256, 0, stream>>>(qkv, ei, alpha, agg, E);

    // ---- combine + LN1 ----
    combine_ln1<<<CDIV(N, 8), 256, 0, stream>>>(x, agg, den, skipf, g1, be1, hbuf, hh, N);

    // ---- FFN GEMMs (gelu fused into GEMM1 epilogue) ----
    wmma_gemm<2><<<dim3(256 / 64, CDIV(N, 128)), 256, 0, stream>>>(hh, w1t, b1, ffa, N, 256, 128, 256);
    wmma_gemm<0><<<dim3(128 / 64, CDIV(N, 128)), 256, 0, stream>>>(ffa, w2t, b2, ff2, N, 128, 256, 128);

    // ---- residual + LN2 -> out ----
    combine_ln2<<<CDIV(N, 8), 256, 0, stream>>>(hbuf, ff2, g2, be2, (float*)d_out, N);
}
